// Attention_26276609917017
// MI455X (gfx1250) — compile-verified, matrix operand order checked by reading, not run on hardware
//
#include <hip/hip_runtime.h>
#include <hip/hip_bf16.h>

// ---------------------------------------------------------------------------
// Attention block for MI455X (gfx1250, wave32, WMMA).
// bf16 operands / f32 accumulate via v_wmma_f32_16x16x32_bf16.
// Tile staging via global_load_async_to_lds_b128 (ASYNCcnt) + double buffering.
// Flash-style streaming softmax (never materializes the 2x16x2048x2048 scores).
// ---------------------------------------------------------------------------

typedef __bf16 bf16_t;
typedef __attribute__((ext_vector_type(16))) __bf16 v16bf;
typedef __attribute__((ext_vector_type(8)))  __bf16 v8bf;
typedef __attribute__((ext_vector_type(8)))  float  v8f;

#define B_    2
#define S_    2048
#define D_    1024
#define H_    16
#define HD_   64
#define NTOK  (B_ * S_)          // 4096
#define LN_EPS 1e-5f
#define LOG_THETA 9.21034037198f // ln(10000)

#if __has_builtin(__builtin_amdgcn_global_load_async_to_lds_b128)
#define ASYNC_LDS 1
typedef int v4i __attribute__((vector_size(16)));
typedef __attribute__((address_space(1))) v4i as1_v4i;  // global b128
typedef __attribute__((address_space(3))) v4i as3_v4i;  // LDS b128
#endif

// 16-byte global -> LDS copy; async (ASYNCcnt-tracked) when available.
static __device__ __forceinline__ void cp16_g2l(const bf16_t* g, bf16_t* l) {
#ifdef ASYNC_LDS
  __builtin_amdgcn_global_load_async_to_lds_b128((as1_v4i*)g, (as3_v4i*)l, 0,
                                                 0);
#else
  *(v8bf*)l = *(const v8bf*)g;
#endif
}

static __device__ __forceinline__ void async_wait_all() {
#ifdef ASYNC_LDS
#if __has_builtin(__builtin_amdgcn_s_wait_asynccnt)
  __builtin_amdgcn_s_wait_asynccnt(0);
#else
  asm volatile("s_wait_asynccnt 0" ::: "memory");
#endif
#endif
}

static __device__ __forceinline__ v8f wmma_bf16(v16bf a, v16bf b, v8f c) {
  return __builtin_amdgcn_wmma_f32_16x16x32_bf16(false, a, false, b, (short)0, c,
                                                 false, false);
}

// ---------------------------------------------------------------------------
// Kernel 1: LayerNorm (fp32 in) -> bf16 out. One row (1024 elems) per block.
// ---------------------------------------------------------------------------
__global__ __launch_bounds__(256) void ln_cast_kernel(
    const float* __restrict__ x, const float* __restrict__ g,
    const float* __restrict__ b, bf16_t* __restrict__ out) {
  const int row  = blockIdx.x;
  const int tid  = threadIdx.x;
  const int lane = tid & 31;
  const int wid  = tid >> 5;
  const float4 v = *(const float4*)(x + (size_t)row * D_ + tid * 4);

  float s  = v.x + v.y + v.z + v.w;
  float s2 = v.x * v.x + v.y * v.y + v.z * v.z + v.w * v.w;
#pragma unroll
  for (int o = 16; o >= 1; o >>= 1) {
    s  += __shfl_xor(s, o, 32);
    s2 += __shfl_xor(s2, o, 32);
  }
  __shared__ float red[16];
  if (lane == 0) { red[wid] = s; red[8 + wid] = s2; }
  __syncthreads();
  if (tid == 0) {
    float ts = 0.f, t2 = 0.f;
#pragma unroll
    for (int i = 0; i < 8; ++i) { ts += red[i]; t2 += red[8 + i]; }
    red[0] = ts; red[1] = t2;
  }
  __syncthreads();
  const float mu   = red[0] * (1.0f / D_);
  const float var  = red[1] * (1.0f / D_) - mu * mu;
  const float rstd = rsqrtf(var + LN_EPS);

  const float xv[4] = {v.x, v.y, v.z, v.w};
#pragma unroll
  for (int k = 0; k < 4; ++k) {
    const int c = tid * 4 + k;
    out[(size_t)row * D_ + c] = (bf16_t)((xv[k] - mu) * rstd * g[c] + b[c]);
  }
}

// ---------------------------------------------------------------------------
// Kernel 2: fp32 -> bf16 cast (weights).
// ---------------------------------------------------------------------------
__global__ __launch_bounds__(256) void cast_bf16_kernel(
    const float* __restrict__ src, bf16_t* __restrict__ dst, int n) {
  const int i = (blockIdx.x * 256 + threadIdx.x) * 4;
  if (i < n) {
    const float4 v = *(const float4*)(src + i);
    dst[i + 0] = (bf16_t)v.x;
    dst[i + 1] = (bf16_t)v.y;
    dst[i + 2] = (bf16_t)v.z;
    dst[i + 3] = (bf16_t)v.w;
  }
}

// ---------------------------------------------------------------------------
// Kernel 3: generic NT GEMM  C[M,N] = A[M,K] * B[N,K]^T (+bias), bf16 in,
// f32 accumulate, mode-switched epilogue.
//   mode 0: bias + full-dim interleaved RoPE on q,k columns -> bf16, ld=N
//   mode 1: bias + scatter to (b,h,s,hd) bf16            (head split)
//   mode 2: bias + scatter to (b,h,hd,s) bf16            (V transposed)
//   mode 3: bias + fp32 output, ld=N                     (final proj)
// Block tile 128x128x32, 8 waves (4x2), each wave 32x64 = 8 WMMA tiles.
// Double-buffered LDS, async tile DMA overlapping WMMA compute.
// ---------------------------------------------------------------------------
#define BM 128
#define BN 128
#define BK 32

__global__ __launch_bounds__(256) void gemm_bf16_nt(
    const bf16_t* __restrict__ A, int lda,
    const bf16_t* __restrict__ Bw, int ldb,
    const float* __restrict__ bias, void* __restrict__ out,
    int M, int N, int K, int mode) {
  __shared__ bf16_t As[2][BM * BK];
  __shared__ bf16_t Bs[2][BN * BK];

  const int tid  = threadIdx.x;
  const int lane = tid & 31;
  const int wid  = tid >> 5;
  const int bm0  = blockIdx.x * BM;
  const int bn0  = blockIdx.y * BN;
  const int wm   = (wid >> 1) * 32;  // wave row in tile
  const int wn   = (wid & 1) * 64;   // wave col in tile

  // per-thread staging pointers (2 x 16B chunks each for A and B), hoisted
  const int c0 = tid, c1 = tid + 256;
  const int ar0 = c0 >> 2, ac0 = (c0 & 3) * 8;
  const int ar1 = c1 >> 2, ac1 = (c1 & 3) * 8;
  const bf16_t* gA0 = A + (size_t)(bm0 + ar0) * lda + ac0;
  const bf16_t* gA1 = A + (size_t)(bm0 + ar1) * lda + ac1;
  const bf16_t* gB0 = Bw + (size_t)(bn0 + ar0) * ldb + ac0;
  const bf16_t* gB1 = Bw + (size_t)(bn0 + ar1) * ldb + ac1;
  const int lo0 = ar0 * BK + ac0, lo1 = ar1 * BK + ac1;

  auto stage = [&](int buf, int k0) {
    cp16_g2l(gA0 + k0, &As[buf][lo0]);
    cp16_g2l(gA1 + k0, &As[buf][lo1]);
    cp16_g2l(gB0 + k0, &Bs[buf][lo0]);
    cp16_g2l(gB1 + k0, &Bs[buf][lo1]);
  };

  v8f acc[2][4];
#pragma unroll
  for (int i = 0; i < 2; ++i)
#pragma unroll
    for (int j = 0; j < 4; ++j) acc[i][j] = (v8f){0, 0, 0, 0, 0, 0, 0, 0};

  const int klo = (lane >> 4) * 16;

  stage(0, 0);
  async_wait_all();
  __syncthreads();

  for (int k0 = 0; k0 < K; k0 += BK) {
    const int buf = (k0 / BK) & 1;
    if (k0 + BK < K) stage(buf ^ 1, k0 + BK);  // async DMA next tile

    v16bf af[2], bfr[4];
#pragma unroll
    for (int sm = 0; sm < 2; ++sm)
      af[sm] = *(const v16bf*)&As[buf][(wm + sm * 16 + (lane & 15)) * BK + klo];
#pragma unroll
    for (int sn = 0; sn < 4; ++sn)
      bfr[sn] = *(const v16bf*)&Bs[buf][(wn + sn * 16 + (lane & 15)) * BK + klo];

#pragma unroll
    for (int sm = 0; sm < 2; ++sm)
#pragma unroll
      for (int sn = 0; sn < 4; ++sn)
        acc[sm][sn] = wmma_bf16(af[sm], bfr[sn], acc[sm][sn]);

    async_wait_all();
    __syncthreads();
  }

  // ---------------- epilogue ----------------
#pragma unroll
  for (int sm = 0; sm < 2; ++sm) {
#pragma unroll
    for (int sn = 0; sn < 4; ++sn) {
#pragma unroll
      for (int j = 0; j < 8; ++j) {
        const int gm = bm0 + wm + sm * 16 + j + ((lane >> 4) << 3);
        const int gn = bn0 + wn + sn * 16 + (lane & 15);
        float v = acc[sm][sn][j] + bias[gn];
        // partner value for RoPE pair (adjacent column == adjacent lane)
        const float partner = __shfl_xor(v, 1, 32);
        if (mode == 0) {
          if (gn < 2 * D_) {  // RoPE on q,k only
            const int dimi = gn & (D_ - 1);
            const int p2   = dimi & ~1;
            const float invf = __expf(-(float)p2 * (LOG_THETA / (float)D_));
            const float ang  = (float)(gm & (S_ - 1)) * invf;
            const float c = __cosf(ang), s = __sinf(ang);
            v = (dimi & 1) ? (v * c + partner * s) : (v * c - partner * s);
          }
          ((bf16_t*)out)[(size_t)gm * N + gn] = (bf16_t)v;
        } else if (mode == 1) {
          const int bb = gm >> 11, ss = gm & (S_ - 1);
          const int hh = gn >> 6,  dd = gn & (HD_ - 1);
          ((bf16_t*)out)[(((size_t)(bb * H_ + hh) * S_ + ss) * HD_) + dd] =
              (bf16_t)v;
        } else if (mode == 2) {
          const int bb = gm >> 11, ss = gm & (S_ - 1);
          const int hh = gn >> 6,  dd = gn & (HD_ - 1);
          ((bf16_t*)out)[(((size_t)(bb * H_ + hh) * HD_ + dd) * S_) + ss] =
              (bf16_t)v;
        } else {
          ((float*)out)[(size_t)gm * N + gn] = v;
        }
      }
    }
  }
}

// ---------------------------------------------------------------------------
// Kernel 4: flash attention per (b,h). qh,kh: (b,h,s,hd) bf16; vT: (b,h,hd,s).
// grid = (S/128, B*H). 8 waves/block, each wave owns 16 query rows.
// K/V tiles (32 keys) staged cooperatively into double-buffered LDS via
// async DMA; S = Q*K^T (2x2 WMMA), online softmax, P*V (4 WMMA).
// ---------------------------------------------------------------------------
#define JT_ (S_ / 32)  // 64 key tiles

__global__ __launch_bounds__(256) void flash_attn_kernel(
    const bf16_t* __restrict__ qh, const bf16_t* __restrict__ kh,
    const bf16_t* __restrict__ vT, const float* __restrict__ mask,
    bf16_t* __restrict__ out) {
  __shared__ bf16_t Ks[2][32 * 64];       // 32 keys x 64 hd  (4 KB each)
  __shared__ bf16_t Vs[2][64 * 32];       // 64 hd x 32 keys  (4 KB each)
  __shared__ bf16_t pshare[8 * 16 * 32];  // per-wave P bounce (C->A layout)

  const int tid  = threadIdx.x;
  const int lane = tid & 31;
  const int wid  = tid >> 5;
  const int bh   = blockIdx.y;           // b*H + h
  const int bidx = bh >> 4;
  const int h    = bh & 15;
  const int q0   = blockIdx.x * 128 + wid * 16;  // this wave's first q row
  const int klo  = (lane >> 4) * 16;
  bf16_t* pmine  = &pshare[wid * (16 * 32)];

  // staging pointers: 1 x 16B K chunk + 1 x 16B V chunk per thread per tile
  const int kr = tid >> 3, kc = (tid & 7) * 8;   // K: 32 rows x 64
  const int vr = tid >> 2, vc = (tid & 3) * 8;   // V: 64 rows x 32
  const bf16_t* gK = kh + ((size_t)bh * S_ + kr) * HD_ + kc;
  const bf16_t* gV = vT + ((size_t)bh * HD_ + vr) * S_ + vc;

  auto stage = [&](int buf, int jt) {
    cp16_g2l(gK + (size_t)jt * 32 * HD_, &Ks[buf][kr * 64 + kc]);
    cp16_g2l(gV + jt * 32, &Vs[buf][vr * 32 + vc]);
  };

  // Q fragments (16 rows x 64 K, two k-steps) -- loaded once
  v16bf qa[2];
#pragma unroll
  for (int t = 0; t < 2; ++t)
    qa[t] = *(const v16bf*)(qh + ((size_t)bh * S_ + q0 + (lane & 15)) * HD_ +
                            t * 32 + klo);

  v8f o[4];
#pragma unroll
  for (int n = 0; n < 4; ++n) o[n] = (v8f){0, 0, 0, 0, 0, 0, 0, 0};
  float mi[8], li[8];
#pragma unroll
  for (int j = 0; j < 8; ++j) { mi[j] = -INFINITY; li[j] = 0.f; }

  stage(0, 0);
  async_wait_all();
  __syncthreads();

  for (int jt = 0; jt < JT_; ++jt) {
    const int buf = jt & 1;
    if (jt + 1 < JT_) stage(buf ^ 1, jt + 1);  // async DMA next K/V tile

    // ---- S tile = Q * K^T  (16 x 32, fp32) ----
    v8f sf[2];
#pragma unroll
    for (int sn = 0; sn < 2; ++sn) {
      v8f a = (v8f){0, 0, 0, 0, 0, 0, 0, 0};
#pragma unroll
      for (int t = 0; t < 2; ++t) {
        const v16bf kb = *(const v16bf*)&Ks[buf][(sn * 16 + (lane & 15)) * 64 +
                                                 t * 32 + klo];
        a = wmma_bf16(qa[t], kb, a);
      }
      sf[sn] = a;
    }
    // ---- scale + mask ----
#pragma unroll
    for (int sn = 0; sn < 2; ++sn) {
#pragma unroll
      for (int j = 0; j < 8; ++j) {
        const int row = q0 + j + ((lane >> 4) << 3);
        const int col = jt * 32 + sn * 16 + (lane & 15);
        sf[sn][j] = sf[sn][j] * 0.125f +
                    mask[((size_t)bidx * S_ + row) * S_ + col];
      }
    }
    // ---- online softmax (row reductions across the 16-lane groups) ----
    float mx[8];
#pragma unroll
    for (int j = 0; j < 8; ++j) {
      mx[j] = fmaxf(sf[0][j], sf[1][j]);
#pragma unroll
      for (int od = 8; od >= 1; od >>= 1)
        mx[j] = fmaxf(mx[j], __shfl_xor(mx[j], od, 32));
    }
    float al[8];
#pragma unroll
    for (int j = 0; j < 8; ++j) {
      const float mn = fmaxf(mi[j], mx[j]);
      al[j] = __expf(mi[j] - mn);
      mi[j] = mn;
    }
#pragma unroll
    for (int sn = 0; sn < 2; ++sn)
#pragma unroll
      for (int j = 0; j < 8; ++j) sf[sn][j] = __expf(sf[sn][j] - mi[j]);
    float rs[8];
#pragma unroll
    for (int j = 0; j < 8; ++j) {
      rs[j] = sf[0][j] + sf[1][j];
#pragma unroll
      for (int od = 8; od >= 1; od >>= 1) rs[j] += __shfl_xor(rs[j], od, 32);
      li[j] = li[j] * al[j] + rs[j];
    }
#pragma unroll
    for (int n = 0; n < 4; ++n)
#pragma unroll
      for (int j = 0; j < 8; ++j) o[n][j] *= al[j];

    // ---- P: C layout -> A layout via per-wave LDS bounce ----
#pragma unroll
    for (int sn = 0; sn < 2; ++sn)
#pragma unroll
      for (int j = 0; j < 8; ++j) {
        const int row = j + ((lane >> 4) << 3);
        const int col = sn * 16 + (lane & 15);
        pmine[row * 32 + col] = (bf16_t)sf[sn][j];
      }
    // same-wave DS ops are in-order; no barrier needed (wave-private region)
    const v16bf pa = *(const v16bf*)(pmine + (lane & 15) * 32 + klo);

    // ---- O += P * V  (V pre-transposed: contraction dim contiguous) ----
#pragma unroll
    for (int n = 0; n < 4; ++n) {
      const v16bf vb =
          *(const v16bf*)&Vs[buf][(n * 16 + (lane & 15)) * 32 + klo];
      o[n] = wmma_bf16(pa, vb, o[n]);
    }

    async_wait_all();
    __syncthreads();
  }

  // ---- finalize: O /= l, scatter back to (b,s,d) bf16 ----
#pragma unroll
  for (int n = 0; n < 4; ++n)
#pragma unroll
    for (int j = 0; j < 8; ++j) {
      const int row = q0 + j + ((lane >> 4) << 3);
      const int ch  = n * 16 + (lane & 15);
      const float v = o[n][j] / li[j];
      out[((size_t)(bidx * S_ + row) * D_) + h * HD_ + ch] = (bf16_t)v;
    }
}

// ---------------------------------------------------------------------------
// Host launcher. Workspace layout (bytes), total ~78 MiB:
//   xn_bf     @ 0        8 MiB   (4096 x 1024 bf16)
//   wqkv_bf   @ 8M       6 MiB   (3072 x 1024)
//   inw_bf    @ 14M      6 MiB
//   outw_bf   @ 20M      2 MiB
//   qkv_bf    @ 22M     24 MiB   (4096 x 3072, RoPE'd q,k + v)
//   qh        @ 46M      8 MiB   (b,h,s,hd)
//   kh        @ 54M      8 MiB
//   vT        @ 62M      8 MiB   (b,h,hd,s)
//   attn_bf   @ 70M      8 MiB   (b,s,d)
// ---------------------------------------------------------------------------
extern "C" void kernel_launch(void* const* d_in, const int* in_sizes, int n_in,
                              void* d_out, int out_size, void* d_ws,
                              size_t ws_size, hipStream_t stream) {
  const float* x     = (const float*)d_in[0];
  const float* mask  = (const float*)d_in[1];
  const float* ln_g  = (const float*)d_in[2];
  const float* ln_b  = (const float*)d_in[3];
  const float* w_qkv = (const float*)d_in[4];
  const float* b_qkv = (const float*)d_in[5];
  const float* in_w  = (const float*)d_in[6];
  const float* in_b  = (const float*)d_in[7];
  const float* out_w = (const float*)d_in[8];
  const float* out_b = (const float*)d_in[9];

  char* ws = (char*)d_ws;
  const size_t MB = 1024ull * 1024ull;
  bf16_t* xn_bf   = (bf16_t*)(ws + 0 * MB);
  bf16_t* wqkv_bf = (bf16_t*)(ws + 8 * MB);
  bf16_t* inw_bf  = (bf16_t*)(ws + 14 * MB);
  bf16_t* outw_bf = (bf16_t*)(ws + 20 * MB);
  bf16_t* qkv_bf  = (bf16_t*)(ws + 22 * MB);
  bf16_t* qh      = (bf16_t*)(ws + 46 * MB);
  bf16_t* kh      = (bf16_t*)(ws + 54 * MB);
  bf16_t* vT      = (bf16_t*)(ws + 62 * MB);
  bf16_t* attn_bf = (bf16_t*)(ws + 70 * MB);

  // 1) LayerNorm + cast
  ln_cast_kernel<<<NTOK, 256, 0, stream>>>(x, ln_g, ln_b, xn_bf);

  // 2) weight casts
  cast_bf16_kernel<<<(3 * D_ * D_) / 1024, 256, 0, stream>>>(w_qkv, wqkv_bf,
                                                             3 * D_ * D_);
  cast_bf16_kernel<<<(3 * D_ * D_) / 1024, 256, 0, stream>>>(in_w, inw_bf,
                                                             3 * D_ * D_);
  cast_bf16_kernel<<<(D_ * D_) / 1024, 256, 0, stream>>>(out_w, outw_bf,
                                                         D_ * D_);

  // 3) fused QKV GEMM + bias + RoPE epilogue (mode 0)
  gemm_bf16_nt<<<dim3(NTOK / BM, (3 * D_) / BN), 256, 0, stream>>>(
      xn_bf, D_, wqkv_bf, D_, b_qkv, qkv_bf, NTOK, 3 * D_, D_, 0);

  // 4) in-projections -> head layouts (modes 1/1/2)
  gemm_bf16_nt<<<dim3(NTOK / BM, D_ / BN), 256, 0, stream>>>(
      qkv_bf + 0 * D_, 3 * D_, inw_bf + 0 * D_ * D_, D_, in_b + 0 * D_, qh,
      NTOK, D_, D_, 1);
  gemm_bf16_nt<<<dim3(NTOK / BM, D_ / BN), 256, 0, stream>>>(
      qkv_bf + 1 * D_, 3 * D_, inw_bf + 1 * D_ * D_, D_, in_b + 1 * D_, kh,
      NTOK, D_, D_, 1);
  gemm_bf16_nt<<<dim3(NTOK / BM, D_ / BN), 256, 0, stream>>>(
      qkv_bf + 2 * D_, 3 * D_, inw_bf + 2 * D_ * D_, D_, in_b + 2 * D_, vT,
      NTOK, D_, D_, 2);

  // 5) flash attention
  flash_attn_kernel<<<dim3(S_ / 128, B_ * H_), 256, 0, stream>>>(qh, kh, vT,
                                                                 mask, attn_bf);

  // 6) output projection -> fp32 d_out
  gemm_bf16_nt<<<dim3(NTOK / BM, D_ / BN), 256, 0, stream>>>(
      attn_bf, D_, outw_bf, D_, out_b, (float*)d_out, NTOK, D_, D_, 3);
}
